// Fingerprint_39230231282148
// MI455X (gfx1250) — matrix operand order repaired and checked by reading, not compiled, not run on hardware
//
#include <hip/hip_runtime.h>
#include <hip/hip_bf16.h>

typedef __attribute__((ext_vector_type(16))) _Float16 v16h;
typedef __attribute__((ext_vector_type(8)))  _Float16 v8h;
typedef __attribute__((ext_vector_type(8)))  float    v8f;

constexpr int BATCH = 256, LEN = 128, NBR = 6, AFD = 39, BFD = 10, NBOND = 256;
constexpr int FD = 200, F3D = 600;
constexpr long BLR  = (long)BATCH * LEN;   // 32768 atom rows
constexpr long BLNR = BLR * NBR;           // 196608 neighbor rows
constexpr int KPS = 64, KPF = 224;         // padded K (mult of 32)
constexpr int NP2 = 256, NP6 = 640;        // padded N (mult of 64 for 16x64 wave tiles)
constexpr float NEGBIG = -9e8f;

enum { EPI_BIAS = 0, EPI_LEAKY = 1, EPI_ELU_RS = 2 };

__device__ __forceinline__ float wave_sum(float v) {
    #pragma unroll
    for (int off = 16; off > 0; off >>= 1) v += __shfl_xor(v, off, 32);
    return v;
}
__device__ __forceinline__ float leakyf(float x) { return x > 0.f ? x : 0.01f * x; }
__device__ __forceinline__ float eluf(float x)   { return x > 0.f ? x : __expf(x) - 1.f; }
__device__ __forceinline__ float sigmf(float x)  { return 1.f / (1.f + __expf(-x)); }

// ---------------- WMMA GEMM: C[M,N] = A[M,Kp](f16) * BT[Np,Kp]^T (f16) + epilogue
// One wave computes a 16x64 output strip: one A fragment feeds 4 WMMAs per K-step.
__global__ __launch_bounds__(32)
void k_wmma_gemm(const _Float16* __restrict__ A, const _Float16* __restrict__ BT,
                 const float* __restrict__ bias, const float* __restrict__ rowscale,
                 float* __restrict__ C, int M, int Nout, int Kp, int mode)
{
    const int tm   = blockIdx.x << 4;
    const int tn   = blockIdx.y << 6;     // 64-wide N strip
    const int lane = threadIdx.x & 31;
    const int kh   = lane >> 4;           // half-wave select
    const int lm   = lane & 15;
    const _Float16* arow = A  + (size_t)(tm + lm) * Kp;
    const _Float16* brow = BT + (size_t)(tn + lm) * Kp;
    const size_t bstride = (size_t)16 * Kp;
    v8f acc[4] = {{}, {}, {}, {}};
    for (int k0 = 0; k0 < Kp; k0 += 32) {
        // A 16x32 f16 fragment: lanes0-15 K=0..15, lanes16-31 K=16..31
        v16h av = *(const v16h*)(arow + k0 + (kh << 4));
        #pragma unroll
        for (int j = 0; j < 4; ++j) {
            // B 32x16 fragment from W^T row (col n): lanes0-15 K=0..15, lanes16-31 K=16..31
            v16h bv = *(const v16h*)(brow + (size_t)j * bstride + k0 + (kh << 4));
            acc[j] = __builtin_amdgcn_wmma_f32_16x16x32_f16(false, av, false, bv,
                                                            (short)0, acc[j], false, false);
        }
    }
    const int mbase = tm + (kh << 3);
    #pragma unroll
    for (int j = 0; j < 4; ++j) {
        const int n = tn + (j << 4) + lm;
        if (n < Nout) {
            const float bv = bias ? bias[n] : 0.f;
            union { v8f v; float f[8]; } u; u.v = acc[j];
            #pragma unroll
            for (int i = 0; i < 8; ++i) {
                float v = u.f[i];
                if (mode == EPI_ELU_RS) v += bv * rowscale[mbase + i];
                else                    v += bv;
                if (mode == EPI_LEAKY)       v = leakyf(v);
                else if (mode == EPI_ELU_RS) v = eluf(v);
                C[(size_t)(mbase + i) * Nout + n] = v;
            }
        }
    }
}

// ---------------- weight transpose+pad+f16:  WT[Np,Kp] = W[K,N]^T
__global__ void k_wtrans(const float* __restrict__ W, _Float16* __restrict__ WT,
                         int K, int Nw, int Kp, int Np)
{
    int idx = blockIdx.x * blockDim.x + threadIdx.x;
    if (idx >= Np * Kp) return;
    int n = idx / Kp, k = idx % Kp;
    float v = (n < Nw && k < K) ? W[(size_t)k * Nw + n] : 0.f;
    WT[idx] = (_Float16)v;
}

// ---------------- activation/pad conversion: dst[rows,Kp] = f16(src[rows,Kin])
__global__ void k_conv(const float* __restrict__ src, _Float16* __restrict__ dst,
                       long rows, int Kin, int Kp)
{
    long idx = (long)blockIdx.x * blockDim.x + threadIdx.x;
    if (idx >= rows * (long)Kp) return;
    long r = idx / Kp; int k = (int)(idx % Kp);
    dst[idx] = (_Float16)(k < Kin ? src[r * (long)Kin + k] : 0.f);
}

// ---------------- gather atom+bond neighbor rows -> concat f16 [BLNR, KPS]
__global__ void k_gather(const float* __restrict__ atoms, const float* __restrict__ bonds,
                         const int* __restrict__ adeg, const int* __restrict__ bdeg,
                         _Float16* __restrict__ dst)
{
    long row = (long)blockIdx.x * blockDim.x + threadIdx.x;
    if (row >= BLNR) return;
    long b = row / ((long)LEN * NBR);
    const float* ar = atoms + (b * LEN + adeg[row]) * (long)AFD;
    const float* br = bonds + (b * NBOND + bdeg[row]) * (long)BFD;
    _Float16* d = dst + row * KPS;
    #pragma unroll
    for (int j = 0; j < AFD; ++j) d[j] = (_Float16)ar[j];
    #pragma unroll
    for (int j = 0; j < BFD; ++j) d[AFD + j] = (_Float16)br[j];
    #pragma unroll
    for (int j = AFD + BFD; j < KPS; ++j) d[j] = (_Float16)0.f;
}

// ---------------- per-atom attention: scores (two dots), softmax over 6, weighted nb sum
__global__ __launch_bounds__(32)
void k_attn(const float* __restrict__ fs,   // [BLR,F] self features
            const float* __restrict__ nf,   // [BLNR,F] materialized neighbors or null
            const float* __restrict__ gsrc, // gather source [BLR,F] (used when nf==null)
            const int* __restrict__ adeg,
            const float* __restrict__ aw,   // [2F]
            const float* __restrict__ abp,  // [1]
            float* __restrict__ wnb, float* __restrict__ sumw)
{
    const long bl = blockIdx.x;
    const int lane = threadIdx.x;
    const long b = bl / LEN;
    const float ab = abp[0];
    const float* fr = fs + bl * FD;
    float part = 0.f;
    for (int f = lane; f < FD; f += 32) part += fr[f] * aw[f];
    const float sa = wave_sum(part);

    const float* nbrp[NBR];
    float sc[NBR], w[NBR];
    bool pad[NBR];
    float mx = -3.4e38f;
    #pragma unroll
    for (int n = 0; n < NBR; ++n) {
        int deg = adeg[bl * NBR + n];
        pad[n] = (deg == LEN - 1);
        nbrp[n] = nf ? (nf + (bl * NBR + n) * (long)FD)
                     : (gsrc + (b * LEN + deg) * (long)FD);
        float p2 = 0.f;
        const float* nr = nbrp[n];
        for (int f = lane; f < FD; f += 32) p2 += nr[f] * aw[FD + f];
        float s = leakyf(sa + wave_sum(p2) + ab);
        if (pad[n]) s += NEGBIG;
        sc[n] = s;
        mx = fmaxf(mx, s);
    }
    float se = 0.f;
    #pragma unroll
    for (int n = 0; n < NBR; ++n) { w[n] = __expf(sc[n] - mx); se += w[n]; }
    float sw = 0.f;
    #pragma unroll
    for (int n = 0; n < NBR; ++n) { w[n] = pad[n] ? 0.f : w[n] / se; sw += w[n]; }
    for (int f = lane; f < FD; f += 32) {
        float acc = 0.f;
        #pragma unroll
        for (int n = 0; n < NBR; ++n) acc += w[n] * nbrp[n][f];
        wnb[bl * FD + f] = acc;
    }
    if (lane == 0) sumw[bl] = sw;
}

// ---------------- GRU gate math (PyTorch GRUCell) on precomputed gi/gh
__global__ void k_gru(const float* __restrict__ gi, const float* __restrict__ gh,
                      const float* __restrict__ h, float* __restrict__ hout,
                      float* __restrict__ act, long rows)
{
    long idx = (long)blockIdx.x * blockDim.x + threadIdx.x;
    if (idx >= rows * FD) return;
    long m = idx / FD; int f = (int)(idx % FD);
    const float* gim = gi + m * (long)F3D;
    const float* ghm = gh + m * (long)F3D;
    float r = sigmf(gim[f]          + ghm[f]);
    float z = sigmf(gim[FD + f]     + ghm[FD + f]);
    float n = tanhf(gim[2 * FD + f] + r * ghm[2 * FD + f]);
    float o = (1.f - z) * n + z * h[idx];
    hout[idx] = o;
    if (act) act[idx] = o > 0.f ? o : 0.f;
}

// ---------------- mol_feature = sum_l activated*mask
__global__ void k_molreduce(const float* __restrict__ act, const float* __restrict__ mask,
                            float* __restrict__ mf)
{
    int idx = blockIdx.x * blockDim.x + threadIdx.x;
    if (idx >= BATCH * FD) return;
    int b = idx / FD, f = idx % FD;
    float s = 0.f;
    for (int l = 0; l < LEN; ++l)
        s += act[((size_t)b * LEN + l) * FD + f] * mask[b * LEN + l];
    mf[idx] = s;
}

// ---------------- out[row] = dot(X[row,:F], v)
__global__ __launch_bounds__(32)
void k_rowdot(const float* __restrict__ X, const float* __restrict__ v,
              float* __restrict__ out)
{
    long row = blockIdx.x; int lane = threadIdx.x;
    const float* xr = X + row * (long)FD;
    float p = 0.f;
    for (int f = lane; f < FD; f += 32) p += xr[f] * v[f];
    p = wave_sum(p);
    if (lane == 0) out[row] = p;
}

// ---------------- per-molecule attention step (softmax over L=128)
__global__ __launch_bounds__(128)
void k_molattn(const float* __restrict__ hcur, const float* __restrict__ snb,
               const float* __restrict__ tr, const float* __restrict__ mask,
               const float* __restrict__ maw, const float* __restrict__ mbp,
               float* __restrict__ ctx)
{
    __shared__ float red[128];
    __shared__ float wsh[128];
    const int b = blockIdx.x, tid = threadIdx.x;
    float p = 0.f;
    for (int f = tid; f < FD; f += 128) {
        float hv = hcur[(size_t)b * FD + f];
        p += (hv > 0.f ? hv : 0.f) * maw[f];     // act_mol = relu(h)
    }
    red[tid] = p; __syncthreads();
    for (int s = 64; s > 0; s >>= 1) { if (tid < s) red[tid] += red[tid + s]; __syncthreads(); }
    const float saself = red[0]; __syncthreads();

    const float m = mask[(size_t)b * LEN + tid];
    float sc = leakyf(saself + snb[(size_t)b * LEN + tid] + mbp[0]);
    if (m == 0.f) sc += NEGBIG;
    red[tid] = sc; __syncthreads();
    for (int s = 64; s > 0; s >>= 1) { if (tid < s) red[tid] = fmaxf(red[tid], red[tid + s]); __syncthreads(); }
    const float mx = red[0]; __syncthreads();
    const float e = __expf(sc - mx);
    red[tid] = e; __syncthreads();
    for (int s = 64; s > 0; s >>= 1) { if (tid < s) red[tid] += red[tid + s]; __syncthreads(); }
    const float tot = red[0]; __syncthreads();
    wsh[tid] = e / tot * m;
    __syncthreads();
    for (int f = tid; f < FD; f += 128) {
        float acc = 0.f;
        for (int l = 0; l < LEN; ++l)
            acc += wsh[l] * tr[((size_t)b * LEN + l) * FD + f];
        ctx[(size_t)b * FD + f] = eluf(acc);
    }
}

// ---------------- prediction head
__global__ __launch_bounds__(32)
void k_pred(const float* __restrict__ mf, const float* __restrict__ ow,
            const float* __restrict__ ob, float* __restrict__ pred)
{
    int b = blockIdx.x, lane = threadIdx.x;
    float p = 0.f;
    for (int f = lane; f < FD; f += 32) p += mf[(size_t)b * FD + f] * ow[f];
    p = wave_sum(p);
    if (lane == 0) pred[b] = p + ob[0];
}

extern "C" void kernel_launch(void* const* d_in, const int* in_sizes, int n_in,
                              void* d_out, int out_size, void* d_ws, size_t ws_size,
                              hipStream_t stream)
{
    (void)in_sizes; (void)n_in; (void)out_size; (void)ws_size;
    const float* atom_list    = (const float*)d_in[0];
    const float* bond_list    = (const float*)d_in[1];
    const int*   adeg         = (const int*)d_in[2];
    const int*   bdeg         = (const int*)d_in[3];
    const float* amask        = (const float*)d_in[4];
    const float* atom_fc_w    = (const float*)d_in[5];
    const float* atom_fc_b    = (const float*)d_in[6];
    const float* nb_fc_w      = (const float*)d_in[7];
    const float* nb_fc_b      = (const float*)d_in[8];
    const float* align_w      = (const float*)d_in[9];
    const float* align_b      = (const float*)d_in[10];
    const float* attend_w     = (const float*)d_in[11];
    const float* attend_b     = (const float*)d_in[12];
    const float* gru_wih      = (const float*)d_in[13];
    const float* gru_whh      = (const float*)d_in[14];
    const float* gru_bih      = (const float*)d_in[15];
    const float* gru_bhh      = (const float*)d_in[16];
    const float* mol_align_w  = (const float*)d_in[17];
    const float* mol_align_b  = (const float*)d_in[18];
    const float* mol_attend_w = (const float*)d_in[19];
    const float* mol_attend_b = (const float*)d_in[20];
    const float* mol_gru_wih  = (const float*)d_in[21];
    const float* mol_gru_whh  = (const float*)d_in[22];
    const float* mol_gru_bih  = (const float*)d_in[23];
    const float* mol_gru_bhh  = (const float*)d_in[24];
    const float* out_w        = (const float*)d_in[25];
    const float* out_b        = (const float*)d_in[26];

    float* out_atom = (float*)d_out;              // [BLR, FD]
    float* out_pred = out_atom + (size_t)BLR * FD;
    float* out_mol  = out_pred + BATCH;

    char* p = (char*)d_ws;
    auto alloc = [&](size_t bytes) -> void* {
        char* r = p; p += (bytes + 255) & ~(size_t)255; return (void*)r;
    };
    float* nf    = (float*)alloc((size_t)BLNR * FD * 4);
    float* hbuf  = (float*)alloc((size_t)BLR * FD * 4);
    float* act   = (float*)alloc((size_t)BLR * FD * 4);
    float* wnb   = (float*)alloc((size_t)BLR * FD * 4);
    float* ctx   = (float*)alloc((size_t)BLR * FD * 4);
    float* gi    = (float*)alloc((size_t)BLR * F3D * 4);
    float* gh    = (float*)alloc((size_t)BLR * F3D * 4);
    float* sumw  = (float*)alloc((size_t)BLR * 4);
    float* snb   = (float*)alloc((size_t)BLR * 4);
    float* tr    = (float*)alloc((size_t)BLR * FD * 4);
    _Float16* A16 = (_Float16*)alloc((size_t)BLNR * KPS * 2); // covers BLR*KPF too
    float* molf  = (float*)alloc((size_t)BATCH * FD * 4);
    float* molh1 = (float*)alloc((size_t)BATCH * FD * 4);
    float* mctx  = (float*)alloc((size_t)BATCH * FD * 4);
    float* mgi   = (float*)alloc((size_t)BATCH * F3D * 4);
    float* mgh   = (float*)alloc((size_t)BATCH * F3D * 4);
    _Float16* WTa   = (_Float16*)alloc((size_t)NP2 * KPS * 2);
    _Float16* WTn   = (_Float16*)alloc((size_t)NP2 * KPS * 2);
    _Float16* WTat0 = (_Float16*)alloc((size_t)NP2 * KPF * 2);
    _Float16* WTat1 = (_Float16*)alloc((size_t)NP2 * KPF * 2);
    _Float16* WTih0 = (_Float16*)alloc((size_t)NP6 * KPF * 2);
    _Float16* WTih1 = (_Float16*)alloc((size_t)NP6 * KPF * 2);
    _Float16* WThh0 = (_Float16*)alloc((size_t)NP6 * KPF * 2);
    _Float16* WThh1 = (_Float16*)alloc((size_t)NP6 * KPF * 2);
    _Float16* WTma  = (_Float16*)alloc((size_t)NP2 * KPF * 2);
    _Float16* WTmi  = (_Float16*)alloc((size_t)NP6 * KPF * 2);
    _Float16* WTmh  = (_Float16*)alloc((size_t)NP6 * KPF * 2);

    auto wtrans = [&](const float* W, _Float16* WT, int K, int Nw, int Kp, int Np) {
        int tot = Np * Kp;
        k_wtrans<<<(tot + 255) / 256, 256, 0, stream>>>(W, WT, K, Nw, Kp, Np);
    };
    auto conv = [&](const float* src, long rows, int Kin, int Kp) {
        long tot = rows * (long)Kp;
        k_conv<<<(unsigned)((tot + 255) / 256), 256, 0, stream>>>(src, A16, rows, Kin, Kp);
    };
    auto gemm = [&](const _Float16* BT, const float* bias, const float* rs, float* C,
                    long M, int Nout, int Kp, int Np, int mode) {
        dim3 g((unsigned)(M / 16), (unsigned)(Np / 64));
        k_wmma_gemm<<<g, 32, 0, stream>>>(A16, BT, bias, rs, C, (int)M, Nout, Kp, mode);
    };

    // ---- weight prep (transpose + pad + f16)
    wtrans(atom_fc_w,            WTa,   AFD,       FD,  KPS, NP2);
    wtrans(nb_fc_w,              WTn,   AFD + BFD, FD,  KPS, NP2);
    wtrans(attend_w,             WTat0, FD,  FD,  KPF, NP2);
    wtrans(attend_w + FD * FD,   WTat1, FD,  FD,  KPF, NP2);
    wtrans(gru_wih,              WTih0, FD,  F3D, KPF, NP6);
    wtrans(gru_wih + FD * F3D,   WTih1, FD,  F3D, KPF, NP6);
    wtrans(gru_whh,              WThh0, FD,  F3D, KPF, NP6);
    wtrans(gru_whh + FD * F3D,   WThh1, FD,  F3D, KPF, NP6);
    wtrans(mol_attend_w,         WTma,  FD,  FD,  KPF, NP2);
    wtrans(mol_gru_wih,          WTmi,  FD,  F3D, KPF, NP6);
    wtrans(mol_gru_whh,          WTmh,  FD,  F3D, KPF, NP6);

    // ---- atom_fc: atom_feature = leaky(atom_list @ W + b)
    conv(atom_list, BLR, AFD, KPS);
    gemm(WTa, atom_fc_b, nullptr, hbuf, BLR, FD, KPS, NP2, EPI_LEAKY);

    // ---- neighbor_fc: leaky(concat(atom_nb, bond_nb) @ W + b)
    k_gather<<<(unsigned)((BLNR + 255) / 256), 256, 0, stream>>>(atom_list, bond_list,
                                                                adeg, bdeg, A16);
    gemm(WTn, nb_fc_b, nullptr, nf, BLNR, FD, KPS, NP2, EPI_LEAKY);

    // ---- radius 0: attention + GRU
    k_attn<<<(unsigned)BLR, 32, 0, stream>>>(hbuf, nf, nullptr, adeg,
                                             align_w, align_b, wnb, sumw);
    conv(wnb, BLR, FD, KPF);
    gemm(WTat0, attend_b, sumw, ctx, BLR, FD, KPF, NP2, EPI_ELU_RS);
    conv(ctx, BLR, FD, KPF);
    gemm(WTih0, gru_bih, nullptr, gi, BLR, F3D, KPF, NP6, EPI_BIAS);
    conv(hbuf, BLR, FD, KPF);
    gemm(WThh0, gru_bhh, nullptr, gh, BLR, F3D, KPF, NP6, EPI_BIAS);
    k_gru<<<(unsigned)((BLR * FD + 255) / 256), 256, 0, stream>>>(gi, gh, hbuf, hbuf, act, BLR);

    // ---- radius 1: attention over gathered activated + GRU (writes final atom_feature)
    k_attn<<<(unsigned)BLR, 32, 0, stream>>>(act, nullptr, act, adeg,
                                             align_w + 2 * FD, align_b + 1, wnb, sumw);
    conv(wnb, BLR, FD, KPF);
    gemm(WTat1, attend_b + FD, sumw, ctx, BLR, FD, KPF, NP2, EPI_ELU_RS);
    conv(ctx, BLR, FD, KPF);
    gemm(WTih1, gru_bih + F3D, nullptr, gi, BLR, F3D, KPF, NP6, EPI_BIAS);
    conv(hbuf, BLR, FD, KPF);
    gemm(WThh1, gru_bhh + F3D, nullptr, gh, BLR, F3D, KPF, NP6, EPI_BIAS);
    k_gru<<<(unsigned)((BLR * FD + 255) / 256), 256, 0, stream>>>(gi, gh, hbuf, out_atom, act, BLR);

    // ---- molecule phase
    k_molreduce<<<(BATCH * FD + 255) / 256, 256, 0, stream>>>(act, amask, molf);
    k_rowdot<<<(unsigned)BLR, 32, 0, stream>>>(act, mol_align_w + FD, snb);
    conv(act, BLR, FD, KPF);
    gemm(WTma, mol_attend_b, nullptr, tr, BLR, FD, KPF, NP2, EPI_BIAS); // loop-invariant

    // step 1
    k_molattn<<<BATCH, 128, 0, stream>>>(molf, snb, tr, amask, mol_align_w, mol_align_b, mctx);
    conv(mctx, BATCH, FD, KPF);
    gemm(WTmi, mol_gru_bih, nullptr, mgi, BATCH, F3D, KPF, NP6, EPI_BIAS);
    conv(molf, BATCH, FD, KPF);
    gemm(WTmh, mol_gru_bhh, nullptr, mgh, BATCH, F3D, KPF, NP6, EPI_BIAS);
    k_gru<<<(BATCH * FD + 255) / 256, 256, 0, stream>>>(mgi, mgh, molf, molh1, nullptr, BATCH);

    // step 2 (writes final mol_feature)
    k_molattn<<<BATCH, 128, 0, stream>>>(molh1, snb, tr, amask, mol_align_w, mol_align_b, mctx);
    conv(mctx, BATCH, FD, KPF);
    gemm(WTmi, mol_gru_bih, nullptr, mgi, BATCH, F3D, KPF, NP6, EPI_BIAS);
    conv(molh1, BATCH, FD, KPF);
    gemm(WTmh, mol_gru_bhh, nullptr, mgh, BATCH, F3D, KPF, NP6, EPI_BIAS);
    k_gru<<<(BATCH * FD + 255) / 256, 256, 0, stream>>>(mgi, mgh, molh1, out_mol, nullptr, BATCH);

    // ---- prediction head
    k_pred<<<BATCH, 32, 0, stream>>>(out_mol, out_w, out_b, out_pred);
}
